// MyModel_87522843559596
// MI455X (gfx1250) — compile-verified, matrix operand order checked by reading, not run on hardware
//
#include <hip/hip_runtime.h>

// ---------------- model constants ----------------
#define T_STEPS   21
#define F_IN      126
#define UNITS     8
#define GATES     32          // 4 * UNITS, gate order i|f|g|o
#define N_CLASSES 27
#define ROWS_PB   16          // batch rows per wave (= WMMA M)
#define ROW_STRIDE (T_STEPS * F_IN)   // 2646 floats per batch row
#define ZOFF      (ROWS_PB * UNITS)   // zeroed LDS row for high-lane reads

typedef __attribute__((ext_vector_type(16))) __bf16 v16bf;
typedef __attribute__((ext_vector_type(2)))  __bf16 bf16x2;
typedef __attribute__((ext_vector_type(8)))  float  v8f;
typedef __attribute__((ext_vector_type(2)))  float  f32x2;

// Assemble a v16bf fragment from 8 packed bf16 pairs (each pair = one 32-bit slot)
union v16u { v16bf v; bf16x2 p[8]; };

static __device__ __forceinline__ float fsig(float x) {
  return __builtin_amdgcn_rcpf(1.0f + __expf(-x));
}

static __device__ __forceinline__ float ftanh(float x) {
#if __has_builtin(__builtin_amdgcn_tanhf)
  return __builtin_amdgcn_tanhf(x);          // gfx1250 v_tanh_f32
#else
  float e = __expf(2.0f * x);
  return 1.0f - 2.0f * __builtin_amdgcn_rcpf(e + 1.0f);
#endif
}

// lane <-> lane^8 exchange: ds_swizzle_b32, group-of-32 mode,
// xor_mask=0x08, or_mask=0, and_mask=0x1f  =>  imm = (8<<10) | 0x1f = 0x201f
static __device__ __forceinline__ float xor8(float x) {
  return __int_as_float(__builtin_amdgcn_ds_swizzle(__float_as_int(x), 0x201f));
}

static __device__ __forceinline__ v8f wmma_bf16(v16bf a, v16bf b, v8f c) {
  // (neg_a, A, neg_b, B, c_mod, C, reuse_a, reuse_b)
  return __builtin_amdgcn_wmma_f32_16x16x32_bf16(false, a, false, b, (short)0, c,
                                                 false, false);
}

// B-fragment of a KxGATES row-major fp32 weight slab, column ncol (0..31);
// rows >= krows are zero-padded. Branch-free clamp+select, pairwise converts.
// Layout mirrors the ISA 16-bit A table with M->N:
//   element j -> k = (j>>3)*16 + half*8 + (j&7)
static __device__ __forceinline__ v16bf load_wfragB(const float* __restrict__ W,
                                                    int krows, int half, int ncol) {
  v16u u;
#pragma unroll
  for (int jp = 0; jp < 8; ++jp) {
    f32x2 d;
#pragma unroll
    for (int e = 0; e < 2; ++e) {
      int j = 2 * jp + e;
      int k = ((j >> 3) << 4) + (half << 3) + (j & 7);
      int ksafe = (k < krows) ? k : 0;
      float v = W[ksafe * GATES + ncol];
      d[e] = (k < krows) ? v : 0.0f;
    }
    u.p[jp] = __builtin_convertvector(d, bf16x2);
  }
  return u.v;
}

// A-fragment (16x32 bf16) of the x tile for k-step kk.
// Validity test split so the always-valid case folds at compile time under
// unrolling (kk,grp,q constants): only (kk=3,grp=1,q=3) keeps a runtime select,
// and it reduces to (half==0). Loads stay unconditional global_load_b64 (NT).
static __device__ __forceinline__ v16bf load_xfragA(const float* __restrict__ xrow,
                                                    int kk, int half) {
  v16u u;
  const f32x2 zz = {0.0f, 0.0f};
#pragma unroll
  for (int grp = 0; grp < 2; ++grp) {
    int kb = kk * 32 + grp * 16 + half * 8;          // multiple of 8 -> 8B aligned
#pragma unroll
    for (int q = 0; q < 4; ++q) {
      int ke = kb + 2 * q;
      bool ok;
      if (kk * 32 + grp * 16 + 8 + 2 * q + 2 <= F_IN)      // worst case (half=1)
        ok = true;                                         // compile-time constant
      else
        ok = (half == 0) && (kk * 32 + grp * 16 + 2 * q + 2 <= F_IN);
      int kesafe = ok ? ke : 0;
      f32x2 d = __builtin_nontemporal_load((const f32x2*)(xrow + kesafe));
      f32x2 s = ok ? d : zz;
      u.p[grp * 4 + q] = __builtin_convertvector(s, bf16x2);
    }
  }
  return u.v;
}

// Gate nonlinearities on the z = [i|f|g|o] D-fragments.
// Only lanes with (lane&8)==0 ever publish h, so compute the "writer"
// interpretation on every lane (dead lanes produce unused garbage):
//   writer's own z0 = i, partner's z0 = f; own z1 = g, partner's z1 = o.
// Sigmoid commutes with the lane exchange, so activate first, then swizzle.
static __device__ __forceinline__ void lstm_gates(v8f z0, v8f z1, float* c, float* h) {
#pragma unroll
  for (int v = 0; v < 8; ++v) {
    float s0 = fsig(z0[v]);        // writers: sig(i); partners: sig(f)
    float gf = xor8(s0);           // writers: sig(f)
    float gg = ftanh(z1[v]);       // writers: tanh(g)
    float s1 = fsig(z1[v]);        // partners: sig(o)
    float go = xor8(s1);           // writers: sig(o)
    c[v] = gf * c[v] + s0 * gg;
    h[v] = go * ftanh(c[v]);
  }
}

// Store writer-lane h values into LDS (16x8 f32) and rebuild the A-fragment for
// the next recurrence: lanes 0..15 element j<8 hold h[m=lane][u=j]; rest zero.
// hp is a per-lane read pointer hoisted out of the loop: lanes 0..15 point at
// their row, lanes 16..31 at a zeroed row -> no per-iteration selects.
// Single-wave workgroup => barriers are hardware NOPs, only compiler ordering.
static __device__ __forceinline__ v16bf store_and_frag(float* __restrict__ hbuf,
                                                       const f32x2* __restrict__ hp,
                                                       const float* h, int lane) {
  const int uu = lane & 7, half = lane >> 4;
  if ((lane & 8) == 0) {
#pragma unroll
    for (int v = 0; v < 8; ++v) hbuf[(v + 8 * half) * UNITS + uu] = h[v];
  }
  __syncthreads();
  v16u a;
#pragma unroll
  for (int jp = 0; jp < 4; ++jp)
    a.p[jp] = __builtin_convertvector(hp[jp], bf16x2);   // ds_load_b64 + pk cvt
  const f32x2 zz = {0.0f, 0.0f};
  const bf16x2 zb = __builtin_convertvector(zz, bf16x2);
#pragma unroll
  for (int jp = 4; jp < 8; ++jp) a.p[jp] = zb;
  __syncthreads();
  return a.v;
}

__global__ __launch_bounds__(32) void lstm2_softmax_kernel(
    const float* __restrict__ x,
    const float* __restrict__ W1k, const float* __restrict__ W1r,
    const float* __restrict__ b1,
    const float* __restrict__ W2k, const float* __restrict__ W2r,
    const float* __restrict__ b2,
    const float* __restrict__ Wd,  const float* __restrict__ bd,
    float* __restrict__ out) {
  __shared__ float hbuf1[ROWS_PB * UNITS + UNITS];   // + zeroed row
  __shared__ float hbuf2[ROWS_PB * UNITS + UNITS];

  const int  lane = threadIdx.x;
  const int  m    = lane & 15;
  const int  half = lane >> 4;
  const size_t row = (size_t)blockIdx.x * ROWS_PB + m;
  const float* xbase = x + row * ROW_STRIDE;

  // zeroed LDS row for high-lane fragment reads (ordered by first barrier)
  if (lane < UNITS) { hbuf1[ZOFF + lane] = 0.0f; hbuf2[ZOFF + lane] = 0.0f; }
  const int roff = (lane < 16) ? lane * UNITS : ZOFF;
  const f32x2* hp1 = (const f32x2*)(hbuf1 + roff);
  const f32x2* hp2 = (const f32x2*)(hbuf2 + roff);

  // ---- preload weight B-fragments (tiny, L2-resident) ----
  v16bf w1kB[4][2];
#pragma unroll
  for (int kk = 0; kk < 4; ++kk)
#pragma unroll
    for (int nt = 0; nt < 2; ++nt)
      w1kB[kk][nt] = load_wfragB(W1k + kk * 32 * GATES, F_IN - kk * 32, half,
                                 m + nt * 16);
  v16bf w1rB[2], w2kB[2], w2rB[2];
#pragma unroll
  for (int nt = 0; nt < 2; ++nt) {
    w1rB[nt] = load_wfragB(W1r, UNITS, half, m + nt * 16);
    w2kB[nt] = load_wfragB(W2k, UNITS, half, m + nt * 16);
    w2rB[nt] = load_wfragB(W2r, UNITS, half, m + nt * 16);
  }
  // bias splat accumulED C vectors, hoisted out of the t-loop
  v8f zb10, zb11, zb20, zb21;
  {
    const float b1lo = b1[m], b1hi = b1[m + 16];
    const float b2lo = b2[m], b2hi = b2[m + 16];
#pragma unroll
    for (int v = 0; v < 8; ++v) {
      zb10[v] = b1lo; zb11[v] = b1hi; zb20[v] = b2lo; zb21[v] = b2hi;
    }
  }

  // ---- on-chip recurrent state ----
  float c1[8], c2[8];
#pragma unroll
  for (int v = 0; v < 8; ++v) { c1[v] = 0.0f; c2[v] = 0.0f; }
  const f32x2 zz = {0.0f, 0.0f};
  const bf16x2 zb = __builtin_convertvector(zz, bf16x2);
  v16u h1A, h2A;
#pragma unroll
  for (int jp = 0; jp < 8; ++jp) { h1A.p[jp] = zb; h2A.p[jp] = zb; }

  for (int t = 0; t < T_STEPS; ++t) {
    const float* xrow = xbase + t * F_IN;
    if (t + 1 < T_STEPS)                              // warm L2 for next slice
      __builtin_prefetch(xrow + F_IN + half * 63, 0, 1);

    // -------- layer 1: z = b1 + x_t @ W1k + h1 @ W1r --------
    v8f z0 = zb10, z1 = zb11;
#pragma unroll
    for (int kk = 0; kk < 4; ++kk) {
      v16bf a = load_xfragA(xrow, kk, half);
      z0 = wmma_bf16(a, w1kB[kk][0], z0);
      z1 = wmma_bf16(a, w1kB[kk][1], z1);
    }
    z0 = wmma_bf16(h1A.v, w1rB[0], z0);
    z1 = wmma_bf16(h1A.v, w1rB[1], z1);

    float h1v[8];
    lstm_gates(z0, z1, c1, h1v);
    h1A.v = store_and_frag(hbuf1, hp1, h1v, lane);

    // -------- layer 2: z = b2 + h1 @ W2k + h2 @ W2r --------
    z0 = zb20; z1 = zb21;
    z0 = wmma_bf16(h1A.v, w2kB[0], z0);
    z1 = wmma_bf16(h1A.v, w2kB[1], z1);
    z0 = wmma_bf16(h2A.v, w2rB[0], z0);
    z1 = wmma_bf16(h2A.v, w2rB[1], z1);

    float h2v[8];
    lstm_gates(z0, z1, c2, h2v);
    h2A.v = store_and_frag(hbuf2, hp2, h2v, lane);
  }

  // -------- dense(27) + softmax on final h2 (fp32 from LDS) --------
  if (lane < 16) {
    float hh[UNITS];
#pragma unroll
    for (int u = 0; u < UNITS; ++u) hh[u] = hbuf2[lane * UNITS + u];
    float logits[N_CLASSES];
    float mx = -3.4e38f;
    for (int cc = 0; cc < N_CLASSES; ++cc) {
      float s = bd[cc];
#pragma unroll
      for (int u = 0; u < UNITS; ++u) s = fmaf(hh[u], Wd[u * N_CLASSES + cc], s);
      logits[cc] = s;
      mx = fmaxf(mx, s);
    }
    float sum = 0.0f;
    for (int cc = 0; cc < N_CLASSES; ++cc) {
      float e = __expf(logits[cc] - mx);
      logits[cc] = e;
      sum += e;
    }
    float inv = __builtin_amdgcn_rcpf(sum);
    float* orow = out + row * N_CLASSES;
    for (int cc = 0; cc < N_CLASSES; ++cc) orow[cc] = logits[cc] * inv;
  }
}

extern "C" void kernel_launch(void* const* d_in, const int* in_sizes, int n_in,
                              void* d_out, int out_size, void* d_ws, size_t ws_size,
                              hipStream_t stream) {
  (void)n_in; (void)out_size; (void)d_ws; (void)ws_size;
  const float* x   = (const float*)d_in[0];
  const float* W1k = (const float*)d_in[1];
  const float* W1r = (const float*)d_in[2];
  const float* b1  = (const float*)d_in[3];
  const float* W2k = (const float*)d_in[4];
  const float* W2r = (const float*)d_in[5];
  const float* b2  = (const float*)d_in[6];
  const float* Wd  = (const float*)d_in[7];
  const float* bd  = (const float*)d_in[8];
  float* out = (float*)d_out;

  const int B = in_sizes[0] / ROW_STRIDE;            // 32768
  dim3 grid((B + ROWS_PB - 1) / ROWS_PB);            // 2048 single-wave blocks
  dim3 block(32);
  lstm2_softmax_kernel<<<grid, block, 0, stream>>>(x, W1k, W1r, b1, W2k, W2r, b2,
                                                   Wd, bd, out);
}
